// PairEnergies_18296560681555
// MI455X (gfx1250) — compile-verified, matrix operand order checked by reading, not run on hardware
//
#include <hip/hip_runtime.h>

#define DEV __device__ __forceinline__

typedef __attribute__((ext_vector_type(16))) __bf16        v16bf;
typedef __attribute__((ext_vector_type(8)))  float         v8f;
typedef __attribute__((ext_vector_type(8)))  unsigned int  v8u;
typedef unsigned int   u32;
typedef unsigned short u16;

namespace {
constexpr int B_ = 2, N_ = 512, K_ = 30, HID_ = 128, L_ = 3, OUT_ = 400, HEADS_ = 4, FF_ = 512;
constexpr int RB_ = B_ * N_ * K_;   // 30720 edge rows
constexpr int RN_ = B_ * N_;        // 1024 node rows
static_assert(RB_ % 256 == 0 && RN_ % 256 == 0, "tile divisibility");
}

DEV u16 f2bf(float f) {
  union { float f; u32 u; } v; v.f = f;
  u32 r = v.u + 0x7FFFu + ((v.u >> 16) & 1u);   // round-to-nearest-even
  return (u16)(r >> 16);
}
DEV float bf2f(u16 s) {
  union { u32 u; float f; } v; v.u = ((u32)s) << 16;
  return v.f;
}
DEV int kOff(int v, int half) {                  // ISA 16-bit A/B fragment K offset
  return ((v < 4) ? (2 * v) : (16 + 2 * (v - 4))) + 8 * half;
}

// ---------------------------------------------------------------------------
// f32 -> bf16 elementwise convert (activation staging)
// ---------------------------------------------------------------------------
__global__ __launch_bounds__(256)
void cvt_f32_bf16_k(const float* __restrict__ s, u16* __restrict__ d, int n) {
  int i = blockIdx.x * blockDim.x + threadIdx.x;
  if (i < n) d[i] = f2bf(s[i]);
}

// ---------------------------------------------------------------------------
// One-time weight pre-pack: W[Ka,Nb] f32 -> per-(chunk,column) fragment order
// so a lane's 8 B-fragment dwords are CONTIGUOUS (=> 2x global_load_b128):
//   Wp[((c*Nb + n)*16) + slot] = pack(W[c*32+ko, n], W[c*32+ko+1, n])
// with slot = half*8 + v, ko = kOff(v, half).
// ---------------------------------------------------------------------------
__global__ __launch_bounds__(256)
void pack_w_k(const float* __restrict__ W, u32* __restrict__ Wp, int Ka, int Nb) {
  const int i = blockIdx.x * blockDim.x + threadIdx.x;
  const int total = (Ka / 32) * Nb * 16;
  if (i >= total) return;
  const int slot = i % 16;
  const int n    = (i / 16) % Nb;
  const int c    = i / (16 * Nb);
  const int half = slot >> 3, v = slot & 7;
  const int k = c * 32 + kOff(v, half);
  Wp[i] = (u32)f2bf(W[(size_t)k * Nb + n]) |
          ((u32)f2bf(W[(size_t)(k + 1) * Nb + n]) << 16);
}

// ---------------------------------------------------------------------------
// bf16 WMMA GEMM, Ka templated (fully unrolled K loop), 4 M-tiles per wave:
// each wave computes a 64x16 output strip; the single B (weight) fragment per
// chunk is reused by 4 v_wmma_f32_16x16x32_bf16 issues.  Both A and B
// fragments are two 16B runs per lane (adjacent for B, 16B gap for A) => paired global_load_b128.
// C/D layout: lane holds column N=lane&15; VGPR v holds row M = v + 8*half.
// ---------------------------------------------------------------------------
template <int KA>
__global__ __launch_bounds__(128)
void gemm_bf16_wmma(const u32* __restrict__ Apk,
                    const u32* __restrict__ Wpk,
                    const float* __restrict__ bias,
                    float* __restrict__ Cf,
                    u16* __restrict__ Cbf,
                    int M, int Nb, int relu)
{
  constexpr int MT = 4;                       // M tiles per wave
  constexpr int NCH = KA / 32;                // k-chunks
  const int lane = threadIdx.x;
  const int wave = blockIdx.x * blockDim.y + threadIdx.y;
  const int m0 = wave * (16 * MT);
  if (m0 >= M) return;
  const int ntile = blockIdx.y;
  const int half = lane >> 4;
  const int l15  = lane & 15;
  const int ncol = ntile * 16 + l15;

  const u32* Ar[MT];
#pragma unroll
  for (int t = 0; t < MT; ++t) {
    Ar[t] = Apk + (size_t)(m0 + t * 16 + l15) * (KA / 2);
    __builtin_prefetch(Ar[t], 0, 0);
  }

  v8f acc[MT];
#pragma unroll
  for (int t = 0; t < MT; ++t) acc[t] = (v8f){0.f, 0.f, 0.f, 0.f, 0.f, 0.f, 0.f, 0.f};

#pragma unroll
  for (int c = 0; c < NCH; ++c) {
    // B (weight) fragment: 8 contiguous dwords per lane (2x b128 loads)
    const u32* wp = Wpk + (((size_t)c * Nb + ncol) << 4) + half * 8;
    v8u bu;
#pragma unroll
    for (int v = 0; v < 8; ++v) bu[v] = wp[v];
    const v16bf bfr = __builtin_bit_cast(v16bf, bu);

#pragma unroll
    for (int t = 0; t < MT; ++t) {
      // A fragment: two runs of 4 contiguous dwords per lane (2x b128 loads)
      v8u au;
#pragma unroll
      for (int v = 0; v < 8; ++v) au[v] = Ar[t][c * 16 + (kOff(v, half) >> 1)];
      const v16bf afr = __builtin_bit_cast(v16bf, au);
      acc[t] = __builtin_amdgcn_wmma_f32_16x16x32_bf16(false, afr, false, bfr,
                                                       (short)0, acc[t], false, false);
    }
  }

  const float bv = bias ? bias[ncol] : 0.f;
#pragma unroll
  for (int t = 0; t < MT; ++t) {
#pragma unroll
    for (int v = 0; v < 8; ++v) {
      const int m = m0 + t * 16 + v + 8 * half;
      float x = acc[t][v] + bv;
      if (relu) x = fmaxf(x, 0.f);
      const size_t off = (size_t)m * Nb + ncol;
      if (Cf)  Cf[off]  = x;
      if (Cbf) Cbf[off] = f2bf(x);
    }
  }
}

// ---------------------------------------------------------------------------
// h_EV = concat(h_V[b, E_idx[b,n,0]], h_V[b, E_idx[b,n,k]], h_E[b,n,k]) (bf16)
// ---------------------------------------------------------------------------
__global__ __launch_bounds__(128)
void gather_hEV_k(const u16* __restrict__ hVb, const u16* __restrict__ hEb,
                  const int* __restrict__ Eix, u16* __restrict__ hEV)
{
  const int row = blockIdx.x;            // b*N*K + n*K + k
  const int d   = threadIdx.x;           // 0..127
  const int bn  = row / K_;
  const int b   = bn / N_;
  const int i0  = Eix[bn * K_];
  const int j   = Eix[row];
  u16* o = hEV + (size_t)row * (3 * HID_);
  o[d]            = hVb[(size_t)(b * N_ + i0) * HID_ + d];
  o[HID_ + d]     = hVb[(size_t)(b * N_ + j)  * HID_ + d];
  o[2 * HID_ + d] = hEb[(size_t)row * HID_ + d];
}

// ---------------------------------------------------------------------------
// Masked multi-head attention over the K=30 neighbors of each (b,n).
// One wave per (b,n,head); lane = query q; all state in VGPRs, fully unrolled.
// ---------------------------------------------------------------------------
__global__ __launch_bounds__(128)
void edge_attn_k(const u16* __restrict__ Qb, const u16* __restrict__ Kb,
                 const u16* __restrict__ Vb, const int* __restrict__ Eix,
                 const float* __restrict__ xm, u16* __restrict__ Ob)
{
  const int q  = threadIdx.x;
  const int h  = threadIdx.y;
  const int bn = blockIdx.x;
  if (q >= K_) return;
  const int b     = bn / N_;
  const int rowb  = bn * K_;
  const float msf = xm[bn];

  float qv[32];
  const u16* qp = Qb + (size_t)(rowb + q) * HID_ + h * 32;
#pragma unroll
  for (int d = 0; d < 32; ++d) qv[d] = bf2f(qp[d]);
  const float mq = xm[b * N_ + Eix[rowb + q]] * msf;

  float lg[K_];
  float mx = -3.4028235e38f;
#pragma unroll
  for (int j = 0; j < K_; ++j) {
    const float mj = xm[b * N_ + Eix[rowb + j]] * msf;
    const u16* kp = Kb + (size_t)(rowb + j) * HID_ + h * 32;
    float acc = 0.f;
#pragma unroll
    for (int t = 0; t < 32; ++t) acc = fmaf(qv[t], bf2f(kp[t]), acc);
    acc *= 0.17677669529663687f;                       // 1/sqrt(d=32)
    lg[j] = (mq * mj > 0.f) ? acc : -3.4028235e38f;
    mx = fmaxf(mx, lg[j]);
  }
  float s = 0.f;
#pragma unroll
  for (int j = 0; j < K_; ++j) { lg[j] = __expf(lg[j] - mx); s += lg[j]; }
  const float inv = 1.f / s;
#pragma unroll
  for (int j = 0; j < K_; ++j) {
    const float mj = xm[b * N_ + Eix[rowb + j]] * msf;
    lg[j] *= inv * (mq * mj);                          // attend = softmax * mask
  }
  float out[32];
#pragma unroll
  for (int d = 0; d < 32; ++d) out[d] = 0.f;
#pragma unroll
  for (int j = 0; j < K_; ++j) {
    const u16* vp = Vb + (size_t)(rowb + j) * HID_ + h * 32;
    const float w = lg[j];
#pragma unroll
    for (int d = 0; d < 32; ++d) out[d] = fmaf(w, bf2f(vp[d]), out[d]);
  }
  u16* op = Ob + (size_t)(rowb + q) * HID_ + h * 32;
#pragma unroll
  for (int d = 0; d < 32; ++d) op[d] = f2bf(out[d]);
}

// ---------------------------------------------------------------------------
// y = LayerNorm(xres + dh) (ddof=1, double-eps like reference) [* mask]
// One wave per 128-wide row; shuffle reduction; emits f32 + bf16 copies.
// ---------------------------------------------------------------------------
__global__ __launch_bounds__(32)
void ln_k(const float* __restrict__ xres, const float* __restrict__ dh,
          const float* __restrict__ g, const float* __restrict__ be,
          const float* __restrict__ mask, int maskDiv,
          float* __restrict__ oF, u16* __restrict__ oB)
{
  const int row  = blockIdx.x;
  const int lane = threadIdx.x;
  const size_t base = (size_t)row * HID_;
  float v[4]; float s = 0.f, ss = 0.f;
#pragma unroll
  for (int i = 0; i < 4; ++i) {
    const int d = lane + 32 * i;
    const float x = xres[base + d] + dh[base + d];
    v[i] = x; s += x; ss += x * x;
  }
#pragma unroll
  for (int o = 16; o > 0; o >>= 1) { s += __shfl_xor(s, o, 32); ss += __shfl_xor(ss, o, 32); }
  const float mu  = s * (1.f / 128.f);
  const float var = (ss - 128.f * mu * mu) * (1.f / 127.f);
  const float sig = sqrtf(var + 1e-6f) + 1e-6f;
  const float msc = mask ? mask[row / maskDiv] : 1.f;
#pragma unroll
  for (int i = 0; i < 4; ++i) {
    const int d = lane + 32 * i;
    const float y = (g[d] * (v[i] - mu) / sig + be[d]) * msc;
    oF[base + d] = y;
    oB[base + d] = f2bf(y);
  }
}

// ---------------------------------------------------------------------------
// merge_duplicate_edges without the NxN scatter: for edge (b,n,k) with j =
// E_idx[b,n,k], find k' (last match) with E_idx[b,j,k'] == n; average.
// ---------------------------------------------------------------------------
__global__ __launch_bounds__(128)
void merge_k(const float* __restrict__ ho, const int* __restrict__ Eix,
             float* __restrict__ out)
{
  const int row = blockIdx.x;
  const int b = row / (N_ * K_);
  const int n = (row / K_) % N_;
  const int j = Eix[row];
  const int baseJ = (b * N_ + j) * K_;
  int krev = -1;
  for (int kk = 0; kk < K_; ++kk) if (Eix[baseJ + kk] == n) krev = kk;
  const size_t r0 = (size_t)row * OUT_;
  const size_t r1 = (krev >= 0) ? ((size_t)(baseJ + krev) * OUT_) : 0;
  for (int c = threadIdx.x; c < OUT_; c += 128) {
    const float rv = (krev >= 0) ? ho[r1 + c] : 0.f;
    out[r0 + c] = 0.5f * (ho[r0 + c] + rv);
  }
}

// ---------------------------------------------------------------------------
extern "C" void kernel_launch(void* const* d_in, const int* in_sizes, int n_in,
                              void* d_out, int out_size, void* d_ws, size_t ws_size,
                              hipStream_t stream)
{
  (void)in_sizes; (void)n_in; (void)out_size; (void)ws_size;

  const float* V    = (const float*)d_in[0];
  const float* E    = (const float*)d_in[1];
  const float* xm   = (const float*)d_in[2];
  const int*   Eix  = (const int*)  d_in[3];
  const float* Wv_w = (const float*)d_in[4];
  const float* Wv_b = (const float*)d_in[5];
  const float* We_w = (const float*)d_in[6];
  const float* We_b = (const float*)d_in[7];
  const float* WQ   = (const float*)d_in[8];
  const float* WK   = (const float*)d_in[9];
  const float* WVm  = (const float*)d_in[10];
  const float* WO   = (const float*)d_in[11];
  const float* ln1g = (const float*)d_in[12];
  const float* ln1b = (const float*)d_in[13];
  const float* ln2g = (const float*)d_in[14];
  const float* ln2b = (const float*)d_in[15];
  const float* ff1w = (const float*)d_in[16];
  const float* ff1b = (const float*)d_in[17];
  const float* ff2w = (const float*)d_in[18];
  const float* ff2b = (const float*)d_in[19];
  const float* Wo_w = (const float*)d_in[20];
  const float* Wo_b = (const float*)d_in[21];
  float* out = (float*)d_out;

  // ---- workspace layout (256B aligned slabs, with aliasing) ----
  char* ws = (char*)d_ws;
  size_t off = 0;
  auto take = [&](size_t bytes) -> char* {
    char* p = ws + off; off += (bytes + 255) & ~(size_t)255; return p;
  };
  // packed weights (bf16-pair dwords), ~1.7 MB total, L2-resident
  u32* Wv_p  = (u32*)take((size_t)HID_ * HID_ * 2);
  u32* We_p  = (u32*)take((size_t)HID_ * HID_ * 2);
  u32* WQ_p  = (u32*)take((size_t)L_ * HID_ * HID_ * 2);
  u32* WK_p  = (u32*)take((size_t)L_ * 3 * HID_ * HID_ * 2);
  u32* WV_p  = (u32*)take((size_t)L_ * 3 * HID_ * HID_ * 2);
  u32* WO_p  = (u32*)take((size_t)L_ * HID_ * HID_ * 2);
  u32* ff1_p = (u32*)take((size_t)L_ * HID_ * FF_ * 2);
  u32* ff2_p = (u32*)take((size_t)L_ * FF_ * HID_ * 2);
  u32* Wo_p  = (u32*)take((size_t)HID_ * OUT_ * 2);
  // activations
  u16*   vin_bf = (u16*)  take((size_t)RN_ * HID_ * 2);
  u16*   hV_bf  = (u16*)  take((size_t)RN_ * HID_ * 2);
  float* hE_f   = (float*)take((size_t)RB_ * HID_ * 4);
  u16*   hE_bf  = (u16*)  take((size_t)RB_ * HID_ * 2);
  char* big = ws + off;                         // start of the aliasable span
  u16*   Q_bf   = (u16*)  take((size_t)RB_ * HID_ * 2);
  u16*   K_bf   = (u16*)  take((size_t)RB_ * HID_ * 2);
  u16*   V_bf   = (u16*)  take((size_t)RB_ * HID_ * 2);
  u16*   hEV_bf = (u16*)  take((size_t)RB_ * 3 * HID_ * 2);
  u16*   at_bf  = (u16*)  take((size_t)RB_ * HID_ * 2);
  float* tmp_f  = (float*)take((size_t)RB_ * HID_ * 4);
  u16*   ff1_bf = hEV_bf;              // aliases hEV+attn (RB*512*2 B exactly)
  u16*   ein_bf = Q_bf;                // E->bf16 staging (dead before Q)
  float* hout   = (float*)big;         // aliases Q..attn span (RB*400*4 B fits)

  auto pack = [&](const float* W, u32* Wp, int Ka, int Nb) {
    const int total = (Ka / 32) * Nb * 16;
    pack_w_k<<<(total + 255) / 256, 256, 0, stream>>>(W, Wp, Ka, Nb);
  };
  auto gemm = [&](const void* A, const u32* Wp, const float* bias,
                  float* Cf, u16* Cbf, int M, int Ka, int Nb, int relu) {
    dim3 grid(M / 256, Nb / 16);
    dim3 blk(32, 4);
    if (Ka == 128)
      gemm_bf16_wmma<128><<<grid, blk, 0, stream>>>((const u32*)A, Wp, bias, Cf, Cbf, M, Nb, relu);
    else if (Ka == 384)
      gemm_bf16_wmma<384><<<grid, blk, 0, stream>>>((const u32*)A, Wp, bias, Cf, Cbf, M, Nb, relu);
    else
      gemm_bf16_wmma<512><<<grid, blk, 0, stream>>>((const u32*)A, Wp, bias, Cf, Cbf, M, Nb, relu);
  };

  // ---- one-time weight packing ----
  pack(Wv_w, Wv_p, HID_, HID_);
  pack(We_w, We_p, HID_, HID_);
  for (int l = 0; l < L_; ++l) {
    pack(WQ  + (size_t)l * HID_ * HID_,     WQ_p  + (size_t)l * HID_ * HID_ / 2,     HID_,     HID_);
    pack(WK  + (size_t)l * 3 * HID_ * HID_, WK_p  + (size_t)l * 3 * HID_ * HID_ / 2, 3 * HID_, HID_);
    pack(WVm + (size_t)l * 3 * HID_ * HID_, WV_p  + (size_t)l * 3 * HID_ * HID_ / 2, 3 * HID_, HID_);
    pack(WO  + (size_t)l * HID_ * HID_,     WO_p  + (size_t)l * HID_ * HID_ / 2,     HID_,     HID_);
    pack(ff1w + (size_t)l * HID_ * FF_,     ff1_p + (size_t)l * HID_ * FF_ / 2,      HID_,     FF_);
    pack(ff2w + (size_t)l * FF_ * HID_,     ff2_p + (size_t)l * FF_ * HID_ / 2,      FF_,      HID_);
  }
  pack(Wo_w, Wo_p, HID_, OUT_);

  // ---- node / edge embeddings ----
  cvt_f32_bf16_k<<<(RN_ * HID_ + 255) / 256, 256, 0, stream>>>(V, vin_bf, RN_ * HID_);
  cvt_f32_bf16_k<<<(RB_ * HID_ + 255) / 256, 256, 0, stream>>>(E, ein_bf, RB_ * HID_);
  gemm(vin_bf, Wv_p, Wv_b, nullptr, hV_bf, RN_, HID_, HID_, 0);
  gemm(ein_bf, We_p, We_b, hE_f,   hE_bf, RB_, HID_, HID_, 0);

  // ---- transformer layers ----
  for (int l = 0; l < L_; ++l) {
    gather_hEV_k<<<RB_, 128, 0, stream>>>(hV_bf, hE_bf, Eix, hEV_bf);
    gemm(hE_bf,  WQ_p + (size_t)l * HID_ * HID_ / 2,     nullptr, nullptr, Q_bf, RB_, HID_,     HID_, 0);
    gemm(hEV_bf, WK_p + (size_t)l * 3 * HID_ * HID_ / 2, nullptr, nullptr, K_bf, RB_, 3 * HID_, HID_, 0);
    gemm(hEV_bf, WV_p + (size_t)l * 3 * HID_ * HID_ / 2, nullptr, nullptr, V_bf, RB_, 3 * HID_, HID_, 0);
    edge_attn_k<<<RN_, dim3(32, HEADS_), 0, stream>>>(Q_bf, K_bf, V_bf, Eix, xm, at_bf);
    gemm(at_bf, WO_p + (size_t)l * HID_ * HID_ / 2, nullptr, tmp_f, nullptr, RB_, HID_, HID_, 0);
    ln_k<<<RB_, 32, 0, stream>>>(hE_f, tmp_f, ln1g + l * HID_, ln1b + l * HID_,
                                 nullptr, K_, hE_f, hE_bf);
    gemm(hE_bf,  ff1_p + (size_t)l * HID_ * FF_ / 2, ff1b + (size_t)l * FF_,
         nullptr, ff1_bf, RB_, HID_, FF_, 1);
    gemm(ff1_bf, ff2_p + (size_t)l * FF_ * HID_ / 2, ff2b + (size_t)l * HID_,
         tmp_f, nullptr, RB_, FF_, HID_, 0);
    ln_k<<<RB_, 32, 0, stream>>>(hE_f, tmp_f, ln2g + l * HID_, ln2b + l * HID_,
                                 xm, K_, hE_f, hE_bf);
  }

  // ---- output projection + reverse-edge merge ----
  gemm(hE_bf, Wo_p, Wo_b, hout, nullptr, RB_, HID_, OUT_, 0);
  merge_k<<<RB_, 128, 0, stream>>>(hout, Eix, out);
}
